// Net3_2396591751560
// MI455X (gfx1250) — matrix operand order repaired and checked by reading, not compiled, not run on hardware
//
#include <hip/hip_runtime.h>

// ---------------------------------------------------------------------------
// GCN: h1 = relu(GCNconv(x,W1,b1)); h2 = relu(GCNconv(h1,W2,b2));
//      out = softmax(h2 @ Wlin + blin)
// GCNconv(improved): agg[c] += xw[r]*dis[r]*dis[c] over edges; + 2*dis[c]^2*xw[c]
// ---------------------------------------------------------------------------

typedef float     v2f  __attribute__((ext_vector_type(2)));
typedef float     v8f  __attribute__((ext_vector_type(8)));
typedef _Float16  v16h __attribute__((ext_vector_type(16)));

#define DIM 128

#if __has_builtin(__builtin_amdgcn_wmma_f32_16x16x4_f32)
#define USE_F32_WMMA 1
#else
#define USE_F32_WMMA 0
#endif

// -------------------------------- utility ----------------------------------
__global__ void zero_f32(float* __restrict__ p, long long n) {
    long long i = (long long)blockIdx.x * blockDim.x + threadIdx.x;
    long long stride = (long long)gridDim.x * blockDim.x;
    for (; i < n; i += stride) p[i] = 0.0f;
}

// deg[col] += 1 per edge
__global__ void deg_count(const long long* __restrict__ ei, float* __restrict__ deg, int E) {
    int e = blockIdx.x * blockDim.x + threadIdx.x;
    if (e < E) atomicAdd(&deg[(int)ei[(size_t)E + e]], 1.0f);
}

// dis[i] = rsqrt(deg[i] + 2.0)   (improved self-loop weight)
__global__ void rsqrt_deg(const float* __restrict__ deg, float* __restrict__ dis, int n) {
    int i = blockIdx.x * blockDim.x + threadIdx.x;
    if (i < n) dis[i] = rsqrtf(deg[i] + 2.0f);
}

// ------------------------------ WMMA GEMM ----------------------------------
// C[M,128] = A[M,K] @ B[K,128].  One block = 16 output rows; wave w owns the
// 16x16 tile at columns w*16.  All 256 lanes active (WMMA needs EXEC==~0).
__global__ __launch_bounds__(256) void gemm_wmma(const float* __restrict__ A,
                                                 const float* __restrict__ B,
                                                 float* __restrict__ C, int K) {
    const int lane = threadIdx.x & 31;
    const int nt   = threadIdx.x >> 5;   // 0..7  -> N tile
    const int mt   = blockIdx.x;         // M tile
    const int idx  = lane & 15;
    const int hi   = lane >> 4;          // 0 (lanes 0-15) / 1 (lanes 16-31)
    const int n    = nt * 16 + idx;
    const float* __restrict__ Arow = A + (size_t)(mt * 16 + idx) * K;

    v8f acc = {0.f, 0.f, 0.f, 0.f, 0.f, 0.f, 0.f, 0.f};

#if USE_F32_WMMA
    // V_WMMA_F32_16X16X4_F32: A lane<16 holds K={0,1}, lane>=16 K={2,3}
    const int koff = hi * 2;
    for (int k0 = 0; k0 < K; k0 += 4) {
        v2f a = *(const v2f*)(Arow + k0 + koff);      // contiguous, 8B aligned
        v2f b;
        b[0] = B[(size_t)(k0 + koff)     * DIM + n];
        b[1] = B[(size_t)(k0 + koff + 1) * DIM + n];
        acc = __builtin_amdgcn_wmma_f32_16x16x4_f32(
            false, a, false, b, (short)0, acc, false, false);
    }
#else
    // fallback: f16 inputs, f32 accumulate (codegen-confirmed builtin)
    for (int k0 = 0; k0 < K; k0 += 32) {
        v16h a, b;
#pragma unroll
        for (int j = 0; j < 8; ++j) {
            a[j]     = (_Float16)Arow[k0 + hi * 8 + j];        // K = hi*8 + j
            a[j + 8] = (_Float16)Arow[k0 + 16 + hi * 8 + j];   // K = 16 + hi*8 + j
        }
#pragma unroll
        for (int j = 0; j < 8; ++j) {
            b[j]     = (_Float16)B[(size_t)(k0 + hi * 8 + j) * DIM + n];
            b[j + 8] = (_Float16)B[(size_t)(k0 + 16 + hi * 8 + j) * DIM + n];
        }
        acc = __builtin_amdgcn_wmma_f32_16x16x32_f16(
            false, a, false, b, (short)0, acc, false, false);
    }
#endif

    // C layout: VGPR r -> row r (lanes 0-15) / row r+8 (lanes 16-31)
    float* __restrict__ Crow = C + (size_t)(mt * 16 + hi * 8) * DIM + n;
#pragma unroll
    for (int r = 0; r < 8; ++r) Crow[(size_t)r * DIM] = acc[r];
}

// --------------------------- edge aggregation ------------------------------
// one wave per edge; lane handles 4 consecutive features (32*4 = 128)
__global__ __launch_bounds__(256) void agg_edges(const float* __restrict__ xw,
                                                 const long long* __restrict__ ei,
                                                 const float* __restrict__ dis,
                                                 float* __restrict__ agg, int E) {
    int wid = blockIdx.x * 8 + (threadIdx.x >> 5);
    if (wid >= E) return;
    int lane = threadIdx.x & 31;
    int r = (int)ei[wid];
    int c = (int)ei[(size_t)E + wid];
    float nrm = dis[r] * dis[c];
    const float4 v = *(const float4*)(xw + (size_t)r * DIM + lane * 4);
    float* dst = agg + (size_t)c * DIM + lane * 4;
    atomicAdd(dst + 0, v.x * nrm);
    atomicAdd(dst + 1, v.y * nrm);
    atomicAdd(dst + 2, v.z * nrm);
    atomicAdd(dst + 3, v.w * nrm);
}

// h = relu(agg + 2*dis^2 * xw + bias), float4 per thread
__global__ void combine_relu(const float* __restrict__ agg, const float* __restrict__ xw,
                             const float* __restrict__ dis, const float* __restrict__ bias,
                             float* __restrict__ h, int ngroups) {
    int g = blockIdx.x * blockDim.x + threadIdx.x;
    if (g >= ngroups) return;
    int row = g >> 5;           // 32 groups of 4 per row of 128
    int cg  = (g & 31) << 2;
    float d = dis[row];
    float s = 2.0f * d * d;
    float4 a = *(const float4*)(agg + (size_t)g * 4);
    float4 x = *(const float4*)(xw  + (size_t)g * 4);
    float4 b = *(const float4*)(bias + cg);
    float4 r;
    r.x = fmaxf(a.x + s * x.x + b.x, 0.0f);
    r.y = fmaxf(a.y + s * x.y + b.y, 0.0f);
    r.z = fmaxf(a.z + s * x.z + b.z, 0.0f);
    r.w = fmaxf(a.w + s * x.w + b.w, 0.0f);
    *(float4*)(h + (size_t)g * 4) = r;
}

// ---------------------- classifier + softmax (8 classes) -------------------
__global__ void cls_softmax(const float* __restrict__ h, const float* __restrict__ Wl,
                            const float* __restrict__ bl, float* __restrict__ out, int n) {
    int i = blockIdx.x * blockDim.x + threadIdx.x;
    if (i >= n) return;
    float acc[8];
#pragma unroll
    for (int c = 0; c < 8; ++c) acc[c] = bl[c];
    const float* hr = h + (size_t)i * DIM;
    for (int k = 0; k < DIM; k += 4) {
        float4 hv = *(const float4*)(hr + k);
#pragma unroll
        for (int c = 0; c < 8; ++c) {
            acc[c] += hv.x * Wl[(k + 0) * 8 + c];
            acc[c] += hv.y * Wl[(k + 1) * 8 + c];
            acc[c] += hv.z * Wl[(k + 2) * 8 + c];
            acc[c] += hv.w * Wl[(k + 3) * 8 + c];
        }
    }
    float m = acc[0];
#pragma unroll
    for (int c = 1; c < 8; ++c) m = fmaxf(m, acc[c]);
    float s = 0.0f;
#pragma unroll
    for (int c = 0; c < 8; ++c) { acc[c] = __expf(acc[c] - m); s += acc[c]; }
    float inv = 1.0f / s;
#pragma unroll
    for (int c = 0; c < 8; ++c) out[(size_t)i * 8 + c] = acc[c] * inv;
}

// ---------------------------------------------------------------------------
extern "C" void kernel_launch(void* const* d_in, const int* in_sizes, int n_in,
                              void* d_out, int out_size, void* d_ws, size_t ws_size,
                              hipStream_t stream) {
    const float*     x  = (const float*)d_in[0];
    const long long* ei = (const long long*)d_in[1];
    const float*     W1 = (const float*)d_in[2];
    const float*     b1 = (const float*)d_in[3];
    const float*     W2 = (const float*)d_in[4];
    const float*     b2 = (const float*)d_in[5];
    const float*     Wl = (const float*)d_in[6];
    const float*     bl = (const float*)d_in[7];
    float* out = (float*)d_out;

    const int N = in_sizes[0] / 512;   // 50000 (multiple of 16)
    const int E = in_sizes[1] / 2;     // 800000
    const size_t ND = (size_t)N * DIM;

    float* xw  = (float*)d_ws;         // [N,128]
    float* agg = xw  + ND;             // [N,128]
    float* h   = agg + ND;             // [N,128]
    float* deg = h   + ND;             // [N]
    float* dis = deg + N;              // [N]

    const int TB = 256;
    // degree + norm
    zero_f32<<<64, TB, 0, stream>>>(deg, N);
    zero_f32<<<2048, TB, 0, stream>>>(agg, (long long)ND);
    deg_count<<<(E + TB - 1) / TB, TB, 0, stream>>>(ei, deg, E);
    rsqrt_deg<<<(N + TB - 1) / TB, TB, 0, stream>>>(deg, dis, N);

    // layer 1: xw = x @ W1 ; aggregate ; h1 = relu(...)
    gemm_wmma<<<N / 16, TB, 0, stream>>>(x, W1, xw, 512);
    agg_edges<<<(E + 7) / 8, TB, 0, stream>>>(xw, ei, dis, agg, E);
    combine_relu<<<(N * 32 + TB - 1) / TB, TB, 0, stream>>>(agg, xw, dis, b1, h, N * 32);

    // layer 2: xw = h1 @ W2 ; aggregate ; h2 = relu(...)  (h overwritten in place)
    zero_f32<<<2048, TB, 0, stream>>>(agg, (long long)ND);
    gemm_wmma<<<N / 16, TB, 0, stream>>>(h, W2, xw, DIM);
    agg_edges<<<(E + 7) / 8, TB, 0, stream>>>(xw, ei, dis, agg, E);
    combine_relu<<<(N * 32 + TB - 1) / TB, TB, 0, stream>>>(agg, xw, dis, b2, h, N * 32);

    // classifier + softmax
    cls_softmax<<<(N + TB - 1) / TB, TB, 0, stream>>>(h, Wl, bl, out, N);
}
